// MixtureOfDepths_79164837200263
// MI455X (gfx1250) — matrix operand order repaired and checked by reading, not compiled, not run on hardware
//
#include <hip/hip_runtime.h>
#include <hip/hip_bf16.h>

// ---------------------------------------------------------------------------
// Mixture-of-Depths for MI455X (gfx1250, wave32, WMMA).
//   B=4, L=4096, D=1024, Dff=4096, k = L*0.5 = 2048.
// ~172 GFLOP vs ~170 MB mandatory HBM traffic (~7.3 us @ 23.3 TB/s): the bf16
// XDL path (v_wmma_f32_16x16x32_bf16) keeps this memory-bound. GEMMs use a
// 2-stage software pipeline (global loads for k+1 in flight during WMMA of k,
// LDS double-buffered, one barrier per K-step). FFN2's A operand (bf16 h in
// workspace) is staged via global_load_async_to_lds_b128 (ASYNCcnt path).
// ---------------------------------------------------------------------------

typedef __bf16 bf16;
typedef __attribute__((ext_vector_type(16))) __bf16 v16bf;
typedef __attribute__((ext_vector_type(8)))  __bf16 v8bf;
typedef __attribute__((ext_vector_type(8)))  float  v8f;
typedef __attribute__((ext_vector_type(4)))  float  v4f;

// problem constants (static per reference)
constexpr int Bsz  = 4;
constexpr int Lseq = 4096;
constexpr int Dmod = 1024;
constexpr int Dff  = 4096;
constexpr int KCAP = 2048;           // k = L * CAPACITY

// GEMM tiling: 128x128 block tile, K-step 32 (= one bf16 WMMA K), 8 waves.
constexpr int BM = 128, BN = 128, BK = 32;
constexpr int LDA_S = BK + 8;        // LDS A row stride (bf16), padded
constexpr int LDB_S = BN + 16;       // LDS B (K-major) row stride, padded

__device__ __forceinline__ void wait_async0()
{
#if __has_builtin(__builtin_amdgcn_s_wait_asynccnt)
    __builtin_amdgcn_s_wait_asynccnt(0);
#else
    asm volatile("s_wait_asynccnt 0x0" ::: "memory");
#endif
}

// ---------------------------------------------------------------------------
// 1) Router scores: one wave per token, lane-strided dot + shuffle reduce.
// ---------------------------------------------------------------------------
__global__ __launch_bounds__(256)
void router_scores(const float* __restrict__ x, const float* __restrict__ wr,
                   float* __restrict__ scores)
{
    const int lane = threadIdx.x & 31;
    const int wave = threadIdx.x >> 5;
    const int tok  = blockIdx.x * 8 + wave;          // B*L tokens total
    const float* xp = x + (size_t)tok * Dmod;
    float s = 0.f;
    #pragma unroll 4
    for (int i = lane; i < Dmod; i += 32) s = fmaf(xp[i], wr[i], s);
    #pragma unroll
    for (int off = 16; off > 0; off >>= 1) s += __shfl_xor(s, off, 32);
    if (lane == 0) scores[tok] = s;
}

// ---------------------------------------------------------------------------
// 2) Top-k per batch via rank counting (matches top_k tie-break: earlier index
//    wins on equal scores), then order-preserving compaction -> sorted indices.
// ---------------------------------------------------------------------------
__global__ __launch_bounds__(1024)
void select_topk(const float* __restrict__ scores, int* __restrict__ top_idx)
{
    __shared__ float          sc[Lseq];
    __shared__ unsigned char  sel[Lseq];
    const int b = blockIdx.x;
    const float* sp = scores + (size_t)b * Lseq;

    for (int i = threadIdx.x; i < Lseq; i += 1024) sc[i] = sp[i];
    __syncthreads();

    for (int i = threadIdx.x; i < Lseq; i += 1024) {
        const float v = sc[i];
        int rank = 0;
        for (int j = 0; j < Lseq; ++j) {
            const float u = sc[j];
            rank += (u > v) || (u == v && j < i);
        }
        sel[i] = (rank < KCAP) ? 1 : 0;
    }
    __syncthreads();

    for (int i = threadIdx.x; i < Lseq; i += 1024) {
        if (sel[i]) {
            int pos = 0;
            for (int j = 0; j < i; ++j) pos += sel[j];
            top_idx[b * KCAP + pos] = i;             // ascending by construction
        }
    }
}

__device__ __forceinline__ float gelu_tanh(float v)
{
    const float t = 0.7978845608028654f * (v + 0.044715f * v * v * v);
    return 0.5f * v * (1.0f + tanhf(t));
}

// ---------------------------------------------------------------------------
// 3) Unified WMMA GEMM: out[m][n] = sum_k A[m][k] * W[n][k]   (torch Linear)
//    MODE 0: A = x (fp32),         W = w_bypass, out -> d_out (fp32)
//    MODE 1: A = x[gather] (fp32), W = w1, gelu, out -> h (bf16 ws)
//    MODE 2: A = h (bf16, async->LDS), W = w2, out scattered onto d_out rows
// A fragment: lane = M + 16g holds K in {8g..8g+7} u {16+8g..16+8g+7}
//             -> two aligned v8bf loads from row-major LDS tile.
// B fragment: lane = N + 16g holds K in the same order as A
//             -> 16 strided bf16 loads from K-major (transposed) LDS tile.
// ---------------------------------------------------------------------------
template <int MODE>
__global__ __launch_bounds__(256)
void mod_gemm(const float* __restrict__ Afp, const bf16* __restrict__ Abf,
              const float* __restrict__ W,   const int*  __restrict__ topidx,
              float* __restrict__ Of,        bf16* __restrict__ Ob)
{
    constexpr int K  = (MODE == 2) ? Dff : Dmod;   // reduction length
    constexpr int NT = (MODE == 1) ? Dff : Dmod;   // output row width
    constexpr int KT = K / BK;

    __shared__ __align__(16) bf16 sA[2][BM * LDA_S];  // row-major  [m][k]
    __shared__ __align__(16) bf16 sB[2][BK * LDB_S];  // K-major    [k][n]

    const int tid  = threadIdx.x;
    const int lane = tid & 31;
    const int wave = tid >> 5;
    const int wm   = (wave >> 1) * 32;             // wave sub-tile: 32 x 64
    const int wn   = (wave & 1) * 64;
    const int mBase = blockIdx.y * BM;
    const int nBase = blockIdx.x * BN;

    // ---- per-thread staging coordinates (K-step invariant, hoisted) ----
    int rA[4], cA[4];
    const float* aRow[4];
    const float* wRow[4];
    #pragma unroll
    for (int i = 0; i < 4; ++i) {
        const int flat = i * 1024 + tid * 4;       // 0..4095 over 128x32 tile
        rA[i] = flat >> 5;                         // tile row (m or n)
        cA[i] = flat & 31;                         // tile col (k)
        wRow[i] = W + (size_t)(nBase + rA[i]) * K;
        if constexpr (MODE == 0) {
            aRow[i] = Afp + (size_t)(mBase + rA[i]) * Dmod;
        } else if constexpr (MODE == 1) {
            const int m = mBase + rA[i];
            const int bb = m >> 11;                              // / KCAP
            aRow[i] = Afp + (size_t)(bb * Lseq + topidx[m]) * Dmod;
        }
    }
    // MODE 2: async-copy coordinates, 2 chunks of 16B per thread (8 KB tile)
    const bf16* hRow[2];
    unsigned    ldsA[2][2];
    int         cH[2];
    if constexpr (MODE == 2) {
        #pragma unroll
        for (int i = 0; i < 2; ++i) {
            const int q = i * 256 + tid;           // 0..511 chunks
            const int r = q >> 2;
            cH[i] = (q & 3) * 8;
            hRow[i] = Abf + (size_t)(mBase + r) * K;
            // low 32 bits of the generic shared pointer == LDS byte address
            ldsA[i][0] = (unsigned)(uintptr_t)&sA[0][r * LDA_S + cH[i]];
            ldsA[i][1] = (unsigned)(uintptr_t)&sA[1][r * LDA_S + cH[i]];
        }
    }

    v4f regA[4], regW[4];

    // issue global loads for K-step kt (A of MODE 2 goes async into stage s)
    auto issue_global = [&](int s, int kt) {
        const int kb = kt * BK;
        if constexpr (MODE == 2) {
            #pragma unroll
            for (int i = 0; i < 2; ++i) {
                const unsigned long long src =
                    (unsigned long long)(uintptr_t)(hRow[i] + kb + cH[i]);
                asm volatile("global_load_async_to_lds_b128 %0, %1, off"
                             :: "v"(ldsA[i][s]), "v"(src) : "memory");
            }
        } else {
            #pragma unroll
            for (int i = 0; i < 4; ++i)
                regA[i] = *(const v4f*)(aRow[i] + kb + cA[i]);
        }
        #pragma unroll
        for (int i = 0; i < 4; ++i)
            regW[i] = *(const v4f*)(wRow[i] + kb + cA[i]);
    };

    // convert staged registers and store into LDS stage s
    auto stage_store = [&](int s) {
        if constexpr (MODE != 2) {
            #pragma unroll
            for (int i = 0; i < 4; ++i) {
                union { bf16 b[4]; uint2 u; } pk;
                pk.b[0] = (bf16)regA[i].x; pk.b[1] = (bf16)regA[i].y;
                pk.b[2] = (bf16)regA[i].z; pk.b[3] = (bf16)regA[i].w;
                *(uint2*)&sA[s][rA[i] * LDA_S + cA[i]] = pk.u;   // ds_store_b64
            }
        }
        #pragma unroll
        for (int i = 0; i < 4; ++i) {                 // transpose W into K-major
            sB[s][(cA[i] + 0) * LDB_S + rA[i]] = (bf16)regW[i].x;
            sB[s][(cA[i] + 1) * LDB_S + rA[i]] = (bf16)regW[i].y;
            sB[s][(cA[i] + 2) * LDB_S + rA[i]] = (bf16)regW[i].z;
            sB[s][(cA[i] + 3) * LDB_S + rA[i]] = (bf16)regW[i].w;
        }
    };

    v8f acc[2][4] = {};
    const int g  = lane >> 4;
    const int lm = lane & 15;

    auto compute = [&](int s) {
        union U16 { v16bf v; v8bf h[2]; };
        U16 a[2], bq[4];
        #pragma unroll
        for (int i = 0; i < 2; ++i) {
            const bf16* pa = &sA[s][(wm + i * 16 + lm) * LDA_S];
            a[i].h[0] = *(const v8bf*)(pa + g * 8);
            a[i].h[1] = *(const v8bf*)(pa + 16 + g * 8);
        }
        #pragma unroll
        for (int j = 0; j < 4; ++j) {
            const bf16* pb = &sB[s][wn + j * 16 + lm];
            #pragma unroll
            for (int e = 0; e < 8; ++e) {
                bq[j].h[0][e] = pb[(g * 8 + e) * LDB_S];
                bq[j].h[1][e] = pb[(16 + g * 8 + e) * LDB_S];
            }
        }
        #pragma unroll
        for (int i = 0; i < 2; ++i)
            #pragma unroll
            for (int j = 0; j < 4; ++j)
                acc[i][j] = __builtin_amdgcn_wmma_f32_16x16x32_bf16(
                    false, a[i].v, false, bq[j].v,
                    (short)0, acc[i][j], false, false);
    };

    // ---- 2-stage pipeline: one barrier per K-step ----
    issue_global(0, 0);
    int s = 0;
    for (int kt = 0; kt < KT; ++kt) {
        stage_store(s);                       // cvt + LDS stores for tile kt
        if constexpr (MODE == 2) wait_async0();  // drain async A-tile writes
        __syncthreads();                      // stage s ready for all waves
        if (kt + 1 < KT) issue_global(s ^ 1, kt + 1);  // overlap with WMMA
        compute(s);
        s ^= 1;
    }

    // ---- epilogue: C layout -> VGPR e holds M = e + 8*(lane>=16), N = lane&15
    const int ln = lane & 15;
    #pragma unroll
    for (int i = 0; i < 2; ++i) {
        #pragma unroll
        for (int j = 0; j < 4; ++j) {
            #pragma unroll
            for (int e = 0; e < 8; ++e) {
                const int m = mBase + wm + i * 16 + e + g * 8;
                const int n = nBase + wn + j * 16 + ln;
                const float v = acc[i][j][e];
                if constexpr (MODE == 0) {
                    Of[(size_t)m * NT + n] = v;
                } else if constexpr (MODE == 1) {
                    Ob[(size_t)m * NT + n] = (bf16)gelu_tanh(v);
                } else {
                    const int bb  = m >> 11;                 // / KCAP
                    const int tok = topidx[m];
                    Of[((size_t)(bb * Lseq + tok)) * NT + n] = v; // overwrite bypass
                }
            }
        }
    }
}

// ---------------------------------------------------------------------------
// Launch: scores -> select -> bypass GEMM -> FFN1(gelu) -> FFN2(scatter).
// Same-stream ordering guarantees the scatter overwrites bypass rows last.
// Workspace: scores 64KB | top_idx 32KB | h (bf16) 64MB.
// ---------------------------------------------------------------------------
extern "C" void kernel_launch(void* const* d_in, const int* in_sizes, int n_in,
                              void* d_out, int out_size, void* d_ws, size_t ws_size,
                              hipStream_t stream)
{
    const float* x        = (const float*)d_in[0];
    const float* w_router = (const float*)d_in[1];
    const float* w_bypass = (const float*)d_in[2];
    const float* w1       = (const float*)d_in[3];
    const float* w2       = (const float*)d_in[4];
    float* out = (float*)d_out;

    float* scores = (float*)d_ws;
    int*   topidx = (int*)((char*)d_ws + (size_t)Bsz * Lseq * sizeof(float));
    bf16*  h      = (bf16*)((char*)d_ws + (size_t)Bsz * Lseq * sizeof(float)
                                        + (size_t)Bsz * KCAP * sizeof(int));

    router_scores<<<dim3(Bsz * Lseq / 8), dim3(256), 0, stream>>>(x, w_router, scores);
    select_topk<<<dim3(Bsz), dim3(1024), 0, stream>>>(scores, topidx);

    // bypass: out = x @ w_bypass^T   (M=16384, N=1024, K=1024)
    mod_gemm<0><<<dim3(Dmod / BN, (Bsz * Lseq) / BM), dim3(256), 0, stream>>>(
        x, nullptr, w_bypass, nullptr, out, nullptr);

    // FFN1: h = gelu(x_sel @ w1^T)   (M=8192, N=4096, K=1024)
    mod_gemm<1><<<dim3(Dff / BN, (Bsz * KCAP) / BM), dim3(256), 0, stream>>>(
        x, nullptr, w1, topidx, nullptr, h);

    // FFN2: out[sel rows] = h @ w2^T (M=8192, N=1024, K=4096), scattered
    mod_gemm<2><<<dim3(Dmod / BN, (Bsz * KCAP) / BM), dim3(256), 0, stream>>>(
        nullptr, h, w2, topidx, out, nullptr);
}
